// MMF_14336600834249
// MI455X (gfx1250) — matrix operand order repaired and checked by reading, not compiled, not run on hardware
//
#include <hip/hip_runtime.h>
#include <hip/hip_bf16.h>

// ---------------- problem constants ----------------
#define BB 16
#define CC 256
#define ICC 64
#define HH 56
#define HW 3136              // 56*56
#define HP 58                // padded output spatial
static __device__ __forceinline__ float lrelu_f(float v) { return v >= 0.f ? v : 0.2f * v; }
#define RSQ 0.99999500003749937f      // rsqrt(1 + 1e-5)
#define ATT_SCALE (1.0f / 896.0f)     // (56*56*256)^-0.5, 896^2 = 802816

// ---------------- WMMA fragment types ----------------
typedef __attribute__((ext_vector_type(16))) __bf16 v16bf;
typedef __attribute__((ext_vector_type(8)))  float  v8f;
typedef __attribute__((ext_vector_type(4)))  unsigned int v4u;

union FragBf { v16bf v; v4u q[2]; unsigned short u[16]; };
union FragF  { v8f   v; float f[8]; };

static __device__ __forceinline__ unsigned short f32_to_bf16(float f) {
  unsigned int u = __builtin_bit_cast(unsigned int, f);
  u += 0x7fffu + ((u >> 16) & 1u);          // round-to-nearest-even
  return (unsigned short)(u >> 16);
}
static __device__ __forceinline__ unsigned int pack_bf16(float lo, float hi) {
  return (unsigned int)f32_to_bf16(lo) | ((unsigned int)f32_to_bf16(hi) << 16);
}
static __device__ __forceinline__ float bf16_to_f32(unsigned short h) {
  unsigned int u = ((unsigned int)h) << 16;
  return __builtin_bit_cast(float, u);
}

// A fragment: 16x32 bf16 tile stored row-major [M][lda].
// lanes 0-15: M=lane, K={0..7,16..23}; lanes 16-31: M=lane-16, K={8..15,24..31}.
static __device__ __forceinline__ FragBf load_a_frag(const unsigned short* src,
                                                     int m_base, int k_base, int lda) {
  FragBf a;
  int lane = threadIdx.x & 31;
  int half = lane >> 4, m = (lane & 15) + m_base;
  const unsigned short* p = src + m * lda + k_base + half * 8;
  a.q[0] = *(const v4u*)(p);         // K = kb+half*8 .. +7
  a.q[1] = *(const v4u*)(p + 16);    // K = kb+16+half*8 .. +7
  return a;
}
// B fragment from an [N][K]-major tile (k contiguous per lane).
static __device__ __forceinline__ FragBf load_bt_frag(const unsigned short* srcT,
                                                      int k_base, int n_base, int ldt) {
  FragBf b;
  int lane = threadIdx.x & 31;
  int half = lane >> 4, n = (lane & 15) + n_base;
  const unsigned short* p = srcT + n * ldt + k_base + half * 8;
  b.q[0] = *(const v4u*)(p);
  b.q[1] = *(const v4u*)(p + 16);
  return b;
}

// ---------------- kernel 1: spatial mean of x_mt ----------------
__global__ __launch_bounds__(256) void mean_kernel(const float* __restrict__ x,
                                                   float* __restrict__ xm) {
  __shared__ float red[256];
  int bc = blockIdx.x;
  size_t base = (size_t)bc * HW;
  float s = 0.f;
  for (int i = threadIdx.x; i < HW; i += 256) s += x[base + i];
  red[threadIdx.x] = s;
  __syncthreads();
  for (int off = 128; off > 0; off >>= 1) {
    if (threadIdx.x < off) red[threadIdx.x] += red[threadIdx.x + off];
    __syncthreads();
  }
  if (threadIdx.x == 0) xm[bc] = red[0] * (1.0f / HW);
}

// ---------------- kernel 2: global-branch q (tiny MLP per batch) ----------------
__global__ __launch_bounds__(256) void globalq_kernel(
    const float* __restrict__ xm, const float* __restrict__ c11_w, const float* __restrict__ c11_b,
    const float* __restrict__ ga_w1, const float* __restrict__ ga_b1,
    const float* __restrict__ ga_g1, const float* __restrict__ ga_be1,
    const float* __restrict__ ga_w2, const float* __restrict__ ga_b2,
    const float* __restrict__ ga_g2, const float* __restrict__ ga_be2,
    float* __restrict__ gq) {
  __shared__ float p[CC];
  __shared__ float h[ICC];
  int b = blockIdx.x, t = threadIdx.x;
  float s = c11_b[t];
  for (int c = 0; c < CC; ++c) s += c11_w[t * CC + c] * xm[b * CC + c];
  p[t] = s;
  __syncthreads();
  if (t < ICC) {
    float s2 = ga_b1[t];
    for (int c = 0; c < CC; ++c) s2 += ga_w1[t * CC + c] * p[c];
    s2 = s2 * (ga_g1[t] * RSQ) + ga_be1[t];
    h[t] = lrelu_f(s2);
  }
  __syncthreads();
  float s3 = ga_b2[t];
  for (int c = 0; c < ICC; ++c) s3 += ga_w2[t * ICC + c] * h[c];
  gq[b * CC + t] = s3 * (ga_g2[t] * RSQ) + ga_be2[t];
}

// ---------------- kernel 3: conv1x1 GEMM (WMMA bf16, double-buffered LDS) ----------------
// Y[b][m,n] = epilogue( sum_k W[m,k] * X[b][k,n] )
// y_mode 0: store bf16 flat [B,M,N]; y_mode 1: store f32 into 58x58 padded interior
#define CONV_LDT 40   // bf16 stride of [N][K] X tile; 40*2=80 bytes, 16B-aligned rows
__global__ __launch_bounds__(128) void conv_gemm_kernel(
    const void* __restrict__ Xv, int x_bf16,
    const float* __restrict__ W, const float* __restrict__ bias,
    const float* __restrict__ bn_g, const float* __restrict__ bn_be, int do_lrelu,
    int M, int K, void* __restrict__ Yv, int y_mode) {
  alignas(16) __shared__ unsigned int lA32[2][64 * 16];            // W tile [m][k] 64x32 bf16
  alignas(16) __shared__ unsigned int lBt32[2][64 * (CONV_LDT/2)]; // X tile [n][k] 64x32 bf16
  const int tid = threadIdx.x, lane = tid & 31, wave = tid >> 5;
  const int n0 = blockIdx.x * 64, m0 = blockIdx.y * 64, b = blockIdx.z;
  const float*          Xf = (const float*)Xv;
  const unsigned short* Xh = (const unsigned short*)Xv;
  const size_t xbase = (size_t)b * K * HW;

  FragF acc[4];
#pragma unroll
  for (int t = 0; t < 4; ++t)
#pragma unroll
    for (int r = 0; r < 8; ++r) acc[t].f[r] = 0.f;

  // register staging for one 64x32 W tile + one 32x64 X tile (8 elems each per thread)
  float          wr0[8], wr1[8];
  float          xr0[8], xr1[8];
  unsigned short xh0[8], xh1[8];

  // ---- prologue: fetch + commit tile for kb = 0 ----
#pragma unroll
  for (int j = 0; j < 8; ++j) {            // W: [m][k] pairs, k contiguous
    int i = tid + j * 128;
    const float* w = &W[(size_t)(m0 + (i >> 4)) * K + 2 * (i & 15)];
    wr0[j] = w[0]; wr1[j] = w[1];
  }
#pragma unroll
  for (int j = 0; j < 8; ++j) {            // X: pairs down k, coalesced along n
    int i = tid + j * 128;
    size_t s0 = xbase + (size_t)(2 * (i >> 6)) * HW + n0 + (i & 63);
    if (x_bf16) { xh0[j] = Xh[s0]; xh1[j] = Xh[s0 + HW]; }
    else        { xr0[j] = Xf[s0]; xr1[j] = Xf[s0 + HW]; }
  }
#pragma unroll
  for (int j = 0; j < 8; ++j) {
    int i = tid + j * 128;
    lA32[0][i] = pack_bf16(wr0[j], wr1[j]);
  }
#pragma unroll
  for (int j = 0; j < 8; ++j) {
    int i = tid + j * 128;
    unsigned int pk = x_bf16 ? ((unsigned int)xh0[j] | ((unsigned int)xh1[j] << 16))
                             : pack_bf16(xr0[j], xr1[j]);
    lBt32[0][(i & 63) * (CONV_LDT / 2) + (i >> 6)] = pk;
  }

  for (int kb = 0; kb < K; kb += 32) {
    const int buf = (kb >> 5) & 1;
    const bool more = (kb + 32) < K;
    __syncthreads();
    // ---- issue next tile's global loads (stay in flight during WMMAs) ----
    if (more) {
#pragma unroll
      for (int j = 0; j < 8; ++j) {
        int i = tid + j * 128;
        const float* w = &W[(size_t)(m0 + (i >> 4)) * K + kb + 32 + 2 * (i & 15)];
        wr0[j] = w[0]; wr1[j] = w[1];
      }
#pragma unroll
      for (int j = 0; j < 8; ++j) {
        int i = tid + j * 128;
        size_t s0 = xbase + (size_t)(kb + 32 + 2 * (i >> 6)) * HW + n0 + (i & 63);
        if (x_bf16) { xh0[j] = Xh[s0]; xh1[j] = Xh[s0 + HW]; }
        else        { xr0[j] = Xf[s0]; xr1[j] = Xf[s0 + HW]; }
      }
    }
    // ---- compute on current buffer: preload all fragments, then 4 WMMAs ----
    FragBf a = load_a_frag((const unsigned short*)lA32[buf], 16 * wave, 0, 32);
    FragBf bq[4];
#pragma unroll
    for (int t = 0; t < 4; ++t)
      bq[t] = load_bt_frag((const unsigned short*)lBt32[buf], 0, t * 16, CONV_LDT);
#pragma unroll
    for (int t = 0; t < 4; ++t)
      acc[t].v = __builtin_amdgcn_wmma_f32_16x16x32_bf16(
          false, a.v, false, bq[t].v, (short)0, acc[t].v, false, false);
    // ---- convert + commit next tile to the other buffer ----
    if (more) {
#pragma unroll
      for (int j = 0; j < 8; ++j) {
        int i = tid + j * 128;
        lA32[buf ^ 1][i] = pack_bf16(wr0[j], wr1[j]);
      }
#pragma unroll
      for (int j = 0; j < 8; ++j) {
        int i = tid + j * 128;
        unsigned int pk = x_bf16 ? ((unsigned int)xh0[j] | ((unsigned int)xh1[j] << 16))
                                 : pack_bf16(xr0[j], xr1[j]);
        lBt32[buf ^ 1][(i & 63) * (CONV_LDT / 2) + (i >> 6)] = pk;
      }
    }
  }

  const int half = lane >> 4, nl = lane & 15, mo = half * 8;
#pragma unroll
  for (int t = 0; t < 4; ++t) {
    int n = n0 + t * 16 + nl;
#pragma unroll
    for (int r = 0; r < 8; ++r) {
      int m = m0 + 16 * wave + mo + r;
      float v = acc[t].f[r];
      if (bias) v += bias[m];
      if (bn_g) v = v * (bn_g[m] * RSQ) + bn_be[m];
      if (do_lrelu) v = lrelu_f(v);
      if (y_mode == 0) {
        ((unsigned short*)Yv)[(size_t)b * M * HW + (size_t)m * HW + n] = f32_to_bf16(v);
      } else {
        int hh = n / HH, ww = n % HH;
        ((float*)Yv)[(((size_t)b * CC + m) * HP + (hh + 1)) * HP + (ww + 1)] = v;
      }
    }
  }
}

// ---------------- kernel 4: fused local+global attention per (b,c) ----------------
#define ATT_LDT 72   // bf16 stride of [N][K] B tile; 72*2=144 bytes, 16B-aligned rows
__global__ __launch_bounds__(128) void attention_kernel(
    const unsigned short* __restrict__ lq, const unsigned short* __restrict__ lk,
    const unsigned short* __restrict__ lv, const unsigned short* __restrict__ gk,
    const unsigned short* __restrict__ gv, const float* __restrict__ gq,
    unsigned short* __restrict__ fout) {
  alignas(16) __shared__ unsigned int sA32[64 * 32];           // A operand [m][k] 64x64 bf16
  alignas(16) __shared__ unsigned int sBt32[64 * (ATT_LDT/2)]; // B operand [n][k] 64x64 bf16
  __shared__ float sS[64 * 65];                                // f32 scores
  const unsigned short* sA  = (const unsigned short*)sA32;
  const unsigned short* sBt = (const unsigned short*)sBt32;
  const int tid = threadIdx.x, lane = tid & 31, wave = tid >> 5;
  const int half = lane >> 4, nl = lane & 15;
  const int bc = blockIdx.x;
  const size_t base = (size_t)bc * HW;
  const float gqv = gq[bc];

  // --- load Q row-major [m][k] and K as [n][k], packed bf16 pairs ---
#pragma unroll
  for (int j = 0; j < 16; ++j) {
    int i = tid + j * 128;
    int r = i >> 5, c = (i & 31) * 2;
    unsigned int pk = 0;
    if (r < HH && c < HH) {
      unsigned short v0 = lq[base + r * HH + c];
      unsigned short v1 = (c + 1 < HH) ? lq[base + r * HH + c + 1] : (unsigned short)0;
      pk = (unsigned int)v0 | ((unsigned int)v1 << 16);
    }
    sA32[i] = pk;
  }
#pragma unroll
  for (int j = 0; j < 16; ++j) {
    int i = tid + j * 128;
    int n = i >> 5, k = (i & 31) * 2;
    unsigned int pk = 0;
    if (n < HH && k < HH) {
      unsigned short v0 = lk[base + n * HH + k];
      unsigned short v1 = (k + 1 < HH) ? lk[base + n * HH + k + 1] : (unsigned short)0;
      pk = (unsigned int)v0 | ((unsigned int)v1 << 16);
    }
    sBt32[n * (ATT_LDT / 2) + (i & 31)] = pk;
  }
  __syncthreads();

  // --- pass 1: scores = (Q @ K^T) * SCALE ---
  FragF s[4];
#pragma unroll
  for (int t = 0; t < 4; ++t)
#pragma unroll
    for (int r = 0; r < 8; ++r) s[t].f[r] = 0.f;
#pragma unroll
  for (int kb = 0; kb < 64; kb += 32) {
    FragBf a = load_a_frag(sA, 16 * wave, kb, 64);
    FragBf bq[4];
#pragma unroll
    for (int t = 0; t < 4; ++t) bq[t] = load_bt_frag(sBt, kb, t * 16, ATT_LDT);
#pragma unroll
    for (int t = 0; t < 4; ++t)
      s[t].v = __builtin_amdgcn_wmma_f32_16x16x32_bf16(
          false, a.v, false, bq[t].v, (short)0, s[t].v, false, false);
  }
#pragma unroll
  for (int t = 0; t < 4; ++t)
#pragma unroll
    for (int r = 0; r < 8; ++r)
      sS[(16 * wave + half * 8 + r) * 65 + t * 16 + nl] = s[t].f[r] * ATT_SCALE;
  __syncthreads();

  // --- row softmax over valid 56 cols -> att (packed bf16) into sA ---
  if (tid < 64) {
    int r = tid;
    float mx = -1e30f;
    for (int c = 0; c < HH; ++c) mx = fmaxf(mx, sS[r * 65 + c]);
    float sum = 0.f;
    for (int c = 0; c < HH; ++c) { float e = __expf(sS[r * 65 + c] - mx); sS[r * 65 + c] = e; sum += e; }
    float inv = 1.f / sum;
#pragma unroll
    for (int c2 = 0; c2 < 32; ++c2) {
      int c = 2 * c2;
      float e0 = (c < HH) ? sS[r * 65 + c] * inv : 0.f;
      float e1 = (c + 1 < HH) ? sS[r * 65 + c + 1] * inv : 0.f;
      sA32[r * 32 + c2] = pack_bf16(e0, e1);
    }
  }
  __syncthreads();

  // --- load V transposed to [n][k] ---
#pragma unroll
  for (int j = 0; j < 16; ++j) {
    int i = tid + j * 128;
    int k2 = i >> 6, n = i & 63, k = 2 * k2;
    unsigned int pk = 0;
    if (n < HH && k < HH) {
      unsigned short v0 = lv[base + k * HH + n];
      unsigned short v1 = (k + 1 < HH) ? lv[base + (k + 1) * HH + n] : (unsigned short)0;
      pk = (unsigned int)v0 | ((unsigned int)v1 << 16);
    }
    sBt32[n * (ATT_LDT / 2) + k2] = pk;
  }
  __syncthreads();

  // --- pass 2: acc = att @ V ---
  FragF acc[4];
#pragma unroll
  for (int t = 0; t < 4; ++t)
#pragma unroll
    for (int r = 0; r < 8; ++r) acc[t].f[r] = 0.f;
#pragma unroll
  for (int kb = 0; kb < 64; kb += 32) {
    FragBf a = load_a_frag(sA, 16 * wave, kb, 64);
    FragBf bq[4];
#pragma unroll
    for (int t = 0; t < 4; ++t) bq[t] = load_bt_frag(sBt, kb, t * 16, ATT_LDT);
#pragma unroll
    for (int t = 0; t < 4; ++t)
      acc[t].v = __builtin_amdgcn_wmma_f32_16x16x32_bf16(
          false, a.v, false, bq[t].v, (short)0, acc[t].v, false, false);
  }
  __syncthreads();

  // --- global branch: gatt = softmax_w(gq * g_k * SCALE) ---
#pragma unroll
  for (int j = 0; j < 32; ++j) {
    int i = tid + j * 128;
    int r = i >> 6, c = i & 63;
    sS[r * 65 + c] = ((r < HH) & (c < HH)) ? bf16_to_f32(gk[base + r * HH + c]) : 0.f;
  }
  __syncthreads();
  if (tid < 64) {
    int r = tid;
    float mx = -1e30f;
    for (int c = 0; c < HH; ++c) mx = fmaxf(mx, gqv * sS[r * 65 + c] * ATT_SCALE);
    float sum = 0.f;
    for (int c = 0; c < HH; ++c) {
      float e = __expf(gqv * sS[r * 65 + c] * ATT_SCALE - mx);
      sS[r * 65 + c] = e; sum += e;
    }
    float inv = 1.f / sum;
#pragma unroll
    for (int c2 = 0; c2 < 32; ++c2) {
      int c = 2 * c2;
      float e0 = (c < HH) ? sS[r * 65 + c] * inv : 0.f;
      float e1 = (c + 1 < HH) ? sS[r * 65 + c + 1] * inv : 0.f;
      sA32[r * 32 + c2] = pack_bf16(e0, e1);
    }
  }
  __syncthreads();
#pragma unroll
  for (int j = 0; j < 16; ++j) {
    int i = tid + j * 128;
    int k2 = i >> 6, n = i & 63, k = 2 * k2;
    unsigned int pk = 0;
    if (n < HH && k < HH) {
      unsigned short v0 = gv[base + k * HH + n];
      unsigned short v1 = (k + 1 < HH) ? gv[base + (k + 1) * HH + n] : (unsigned short)0;
      pk = (unsigned int)v0 | ((unsigned int)v1 << 16);
    }
    sBt32[n * (ATT_LDT / 2) + k2] = pk;
  }
  __syncthreads();

  // --- pass 3: acc += gatt @ g_v ---
#pragma unroll
  for (int kb = 0; kb < 64; kb += 32) {
    FragBf a = load_a_frag(sA, 16 * wave, kb, 64);
    FragBf bq[4];
#pragma unroll
    for (int t = 0; t < 4; ++t) bq[t] = load_bt_frag(sBt, kb, t * 16, ATT_LDT);
#pragma unroll
    for (int t = 0; t < 4; ++t)
      acc[t].v = __builtin_amdgcn_wmma_f32_16x16x32_bf16(
          false, a.v, false, bq[t].v, (short)0, acc[t].v, false, false);
  }

  // --- store f = l_fuse + g_fuse (bf16, valid 56x56) ---
#pragma unroll
  for (int t = 0; t < 4; ++t) {
    int n = t * 16 + nl;
#pragma unroll
    for (int r = 0; r < 8; ++r) {
      int m = 16 * wave + half * 8 + r;
      if (m < HH && n < HH) fout[base + m * HH + n] = f32_to_bf16(acc[t].f[r]);
    }
  }
}

// ---------------- kernel 5: pre-fill output with bias (covers the pad ring) ----------------
__global__ __launch_bounds__(256) void fill_out_kernel(float* __restrict__ out,
                                                       const float* __restrict__ c1_b) {
  int bo = blockIdx.x;                       // b*256 + o
  float v = c1_b[bo & (CC - 1)];
  size_t base = (size_t)bo * HP * HP;
  for (int i = threadIdx.x; i < HP * HP; i += 256) out[base + i] = v;
}

// ---------------- launch ----------------
extern "C" void kernel_launch(void* const* d_in, const int* in_sizes, int n_in,
                              void* d_out, int out_size, void* d_ws, size_t ws_size,
                              hipStream_t stream) {
  const float* x_s   = (const float*)d_in[0];
  const float* x_fq  = (const float*)d_in[1];
  const float* x_mt  = (const float*)d_in[2];
  const float* la_w1 = (const float*)d_in[3];
  const float* la_b1 = (const float*)d_in[4];
  const float* la_g1 = (const float*)d_in[5];
  const float* la_be1= (const float*)d_in[6];
  const float* la_w2 = (const float*)d_in[7];
  const float* la_b2 = (const float*)d_in[8];
  const float* la_g2 = (const float*)d_in[9];
  const float* la_be2= (const float*)d_in[10];
  const float* lk_w  = (const float*)d_in[11];
  const float* lk_b  = (const float*)d_in[12];
  const float* lv_w  = (const float*)d_in[13];
  const float* ga_w1 = (const float*)d_in[14];
  const float* ga_b1 = (const float*)d_in[15];
  const float* ga_g1 = (const float*)d_in[16];
  const float* ga_be1= (const float*)d_in[17];
  const float* ga_w2 = (const float*)d_in[18];
  const float* ga_b2 = (const float*)d_in[19];
  const float* ga_g2 = (const float*)d_in[20];
  const float* ga_be2= (const float*)d_in[21];
  const float* gk_w  = (const float*)d_in[22];
  const float* gk_b  = (const float*)d_in[23];
  const float* gv_w  = (const float*)d_in[24];
  const float* c11_w = (const float*)d_in[25];
  const float* c11_b = (const float*)d_in[26];
  const float* c1_w  = (const float*)d_in[27];
  const float* c1_b  = (const float*)d_in[28];

  char* ws = (char*)d_ws;
  size_t off = 0;
  auto alloc = [&](size_t bytes) -> void* {
    void* p = ws + off;
    off += (bytes + 255) & ~(size_t)255;
    return p;
  };
  float*          xm  = (float*)alloc((size_t)BB * CC * 4);
  float*          gq  = (float*)alloc((size_t)BB * CC * 4);
  unsigned short* h1  = (unsigned short*)alloc((size_t)BB * ICC * HW * 2);
  unsigned short* lqb = (unsigned short*)alloc((size_t)BB * CC * HW * 2);
  unsigned short* lkb = (unsigned short*)alloc((size_t)BB * CC * HW * 2);
  unsigned short* lvb = (unsigned short*)alloc((size_t)BB * CC * HW * 2);
  unsigned short* gkb = (unsigned short*)alloc((size_t)BB * CC * HW * 2);
  unsigned short* gvb = (unsigned short*)alloc((size_t)BB * CC * HW * 2);
  unsigned short* fb  = (unsigned short*)alloc((size_t)BB * CC * HW * 2);

  mean_kernel<<<BB * CC, 256, 0, stream>>>(x_mt, xm);
  globalq_kernel<<<BB, 256, 0, stream>>>(xm, c11_w, c11_b, ga_w1, ga_b1, ga_g1, ga_be1,
                                         ga_w2, ga_b2, ga_g2, ga_be2, gq);

  dim3 gThin(HW / 64, ICC / 64, BB);   // 49 x 1 x 16
  dim3 gFull(HW / 64, CC / 64, BB);    // 49 x 4 x 16
  // la path: h1 = lrelu(bn(W1 x_s + b1)); l_q = bn(W2 h1 + b2)
  conv_gemm_kernel<<<gThin, 128, 0, stream>>>(x_s, 0, la_w1, la_b1, la_g1, la_be1, 1,
                                              ICC, CC, h1, 0);
  conv_gemm_kernel<<<gFull, 128, 0, stream>>>(h1, 1, la_w2, la_b2, la_g2, la_be2, 0,
                                              CC, ICC, lqb, 0);
  conv_gemm_kernel<<<gFull, 128, 0, stream>>>(x_s, 0, lk_w, lk_b, nullptr, nullptr, 0,
                                              CC, CC, lkb, 0);
  conv_gemm_kernel<<<gFull, 128, 0, stream>>>(x_s, 0, lv_w, nullptr, nullptr, nullptr, 0,
                                              CC, CC, lvb, 0);
  conv_gemm_kernel<<<gFull, 128, 0, stream>>>(x_fq, 0, gk_w, gk_b, nullptr, nullptr, 0,
                                              CC, CC, gkb, 0);
  conv_gemm_kernel<<<gFull, 128, 0, stream>>>(x_fq, 0, gv_w, nullptr, nullptr, nullptr, 0,
                                              CC, CC, gvb, 0);

  attention_kernel<<<BB * CC, 128, 0, stream>>>(lqb, lkb, lvb, gkb, gvb, gq, fb);

  fill_out_kernel<<<BB * CC, 256, 0, stream>>>((float*)d_out, c1_b);
  conv_gemm_kernel<<<gFull, 128, 0, stream>>>(fb, 1, c1_w, c1_b, nullptr, nullptr, 0,
                                              CC, CC, d_out, 1);
}
